// GTBlock_46909632807104
// MI455X (gfx1250) — matrix operand order repaired and checked by reading, not compiled, not run on hardware
//
#include <hip/hip_runtime.h>
#include <hip/hip_bf16.h>
#include <math.h>

#define CDIM  128
#define NHEAD 4

typedef __bf16 bf16_t;
typedef bf16_t v16bf __attribute__((ext_vector_type(16)));
typedef bf16_t v8bf  __attribute__((ext_vector_type(8)));
typedef float  v8f   __attribute__((ext_vector_type(8)));

// ---------------------------------------------------------------- utilities
__device__ __forceinline__ float gelu_exact(float x) {
    return 0.5f * x * (1.0f + erff(x * 0.70710678118654752440f));
}
// order-preserving int encoding of float for atomicMax-based segment max
__device__ __forceinline__ int enc_f32(float f) {
    int i = __float_as_int(f);
    return (i < 0) ? (i ^ 0x7FFFFFFF) : i;
}
__device__ __forceinline__ float dec_f32(int i) {
    if (i == (int)0x80000000) return 0.0f;  // sentinel: node had no edges of this relation
    int j = (i < 0) ? (i ^ 0x7FFFFFFF) : i;
    return __int_as_float(j);
}
__device__ __forceinline__ v16bf cat16(v8bf lo, v8bf hi) {
    return __builtin_shufflevector(lo, hi, 0,1,2,3,4,5,6,7,8,9,10,11,12,13,14,15);
}

// ---------------------------------------------------------------- WMMA GEMM
// out[nrows x 128] = act( A[nrows x 128] @ W[128 x 128] + bias )
// A is bf16 row-major. Bt is the bf16 *transposed* weight, Bt[n][k] = W[k][n].
// The whole 128x128 bf16 weight (32 KB) is staged in LDS once per block;
// all 8 waves then feed WMMA B-fragments from ds_load_b128 (low latency,
// 8x less L2 traffic than per-wave global fetches).
// Each wave owns 16 rows x 128 cols (8 tiles of 16x16), 8 waves per block.
// act: 0=none 1=exact-gelu; outmode: 0=store f32, 1=store bf16, 2=accumulate f32
__global__ void __launch_bounds__(256)
gemm_bf16_wmma(const bf16_t* __restrict__ A, const bf16_t* __restrict__ Bt,
               const float* __restrict__ bias, float* __restrict__ outF,
               bf16_t* __restrict__ outB, int nrows, int act, int outmode)
{
    __shared__ bf16_t ldsB[CDIM * CDIM];          // 32 KB of the WGP's 320 KB LDS

    // cooperative coalesced copy: 2048 x uint4, 8 per thread
    {
        const uint4* src = (const uint4*)Bt;
        uint4*       dst = (uint4*)ldsB;
#pragma unroll
        for (int i = 0; i < 8; ++i)
            dst[threadIdx.x + i * 256] = src[threadIdx.x + i * 256];
    }
    __syncthreads();

    const int lane    = threadIdx.x & 31;
    const int rowTile = blockIdx.x * 8 + (threadIdx.x >> 5);
    const int rowBase = rowTile << 4;
    if (rowBase >= nrows) return;            // wave-uniform: EXEC stays full for WMMA

    int mrow = rowBase + (lane & 15);
    if (mrow >= nrows) mrow = nrows - 1;     // clamp (results discarded on store)
    // A-frag (16x32 bf16): lanes 0-15 hold K {0-7,16-23}, lanes 16-31 hold K {8-15,24-31}
    const bf16_t* aRow = A + (size_t)mrow * CDIM + ((lane >> 4) << 3);

    const int ncol  = lane & 15;             // B-frag: lane = output column within tile
    const int bkOff = (lane >> 4) << 3;      // lanes 0-15: K {0-7,16-23}, lanes 16-31: K {8-15,24-31}

    // preload all four A fragments (one clause of 8 global_load_b128)
    v16bf afr[4];
#pragma unroll
    for (int ks = 0; ks < 4; ++ks) {
        const bf16_t* ap = aRow + ks * 32;
        afr[ks] = cat16(*(const v8bf*)ap, *(const v8bf*)(ap + 16));
    }

    v8f acc[8];
#pragma unroll
    for (int i = 0; i < 8; ++i)
#pragma unroll
        for (int j = 0; j < 8; ++j) acc[i][j] = 0.0f;

#pragma unroll
    for (int ct = 0; ct < 8; ++ct) {         // 8 column tiles of 16
        const bf16_t* bcol = ldsB + (size_t)(ct * 16 + ncol) * CDIM + bkOff;
#pragma unroll
        for (int ks = 0; ks < 4; ++ks) {     // K = 128 = 4 x 32
            v16bf bfr = cat16(*(const v8bf*)(bcol + ks * 32),
                              *(const v8bf*)(bcol + ks * 32 + 16));
            acc[ct] = __builtin_amdgcn_wmma_f32_16x16x32_bf16(
                false, afr[ks], false, bfr, (short)0, acc[ct], false, false);
        }
    }

    // C/D layout: lane%16 = column, row = rowBase + vgpr + 8*(lane/16)
    const int rOff = (lane >> 4) << 3;
#pragma unroll
    for (int ct = 0; ct < 8; ++ct) {
        const int col = ct * 16 + ncol;
        const float bv = bias[col];
#pragma unroll
        for (int v = 0; v < 8; ++v) {
            const int row = rowBase + rOff + v;
            if (row >= nrows) continue;
            float val = acc[ct][v] + bv;
            if (act) val = gelu_exact(val);
            const size_t idx = (size_t)row * CDIM + col;
            if (outmode == 0)      outF[idx] = val;
            else if (outmode == 1) outB[idx] = (bf16_t)val;
            else                   outF[idx] += val;
        }
    }
}

// ---------------------------------------------------------------- LayerNorm -> bf16
// one wave per row; lane l handles channels 4l..4l+3
__global__ void __launch_bounds__(256)
ln_to_bf16(const float* __restrict__ x, const float* __restrict__ w,
           const float* __restrict__ b, bf16_t* __restrict__ x1b, int nrows)
{
    const int lane = threadIdx.x & 31;
    const int row  = blockIdx.x * 8 + (threadIdx.x >> 5);
    if (row >= nrows) return;
    const int c = lane << 2;
    const float4 v = *(const float4*)(x + (size_t)row * CDIM + c);
    float s  = v.x + v.y + v.z + v.w;
    float sq = v.x*v.x + v.y*v.y + v.z*v.z + v.w*v.w;
#pragma unroll
    for (int m = 16; m >= 1; m >>= 1) {
        s  += __shfl_xor(s,  m, 32);
        sq += __shfl_xor(sq, m, 32);
    }
    const float mu  = s * (1.0f / 128.0f);
    const float var = sq * (1.0f / 128.0f) - mu * mu;
    const float rs  = rsqrtf(var + 1e-5f);
    const float4 wv = *(const float4*)(w + c);
    const float4 bv = *(const float4*)(b + c);
    bf16_t* o = x1b + (size_t)row * CDIM + c;
    o[0] = (bf16_t)((v.x - mu) * rs * wv.x + bv.x);
    o[1] = (bf16_t)((v.y - mu) * rs * wv.y + bv.y);
    o[2] = (bf16_t)((v.z - mu) * rs * wv.z + bv.z);
    o[3] = (bf16_t)((v.w - mu) * rs * wv.w + bv.w);
}

// ---------------------------------------------------------------- small helpers
__global__ void fill_f32(float* __restrict__ p, float val, int n) {
    int i = blockIdx.x * 256 + threadIdx.x;
    if (i < n) p[i] = val;
}
__global__ void fill_i32(int* __restrict__ p, int val, int n) {
    int i = blockIdx.x * 256 + threadIdx.x;
    if (i < n) p[i] = val;
}
// transpose one 128x128 f32 weight to bf16: Wt[n][k] = W[k][n]
__global__ void wt_prep(const float* __restrict__ W, bf16_t* __restrict__ Wt) {
    int i = blockIdx.x * 256 + threadIdx.x;   // 0..16383
    int n = i >> 7, k = i & 127;
    Wt[i] = (bf16_t)W[k * CDIM + n];
}

// ---------------------------------------------------------------- edge kernels
// E1: wave per edge. lane l -> channels 4l..4l+3, head = l/8.
// e[h] = sum_d att[h][d] * leakyrelu(xl[src]+xr[dst]); atomicMax into emax[dst][h]
__global__ void __launch_bounds__(256)
edge_logits(const long long* __restrict__ ei, const int* __restrict__ ea,
            const float* __restrict__ xl, const float* __restrict__ xr,
            const float* __restrict__ att, float* __restrict__ ebuf,
            int* __restrict__ emax, int rel, int nedges)
{
    const int lane = threadIdx.x & 31;
    const int e    = blockIdx.x * 8 + (threadIdx.x >> 5);
    if (e >= nedges) return;
    if (ea[e] + 1 != rel) return;
    const long long s = ei[e];
    const long long d = ei[nedges + e];
    const int c = lane << 2;
    const float4 a  = *(const float4*)(xl + (size_t)s * CDIM + c);
    const float4 r  = *(const float4*)(xr + (size_t)d * CDIM + c);
    const float4 at = *(const float4*)(att + c);   // att pre-offset by rel*128 (H*DH == 128)
    float p = 0.0f, z;
    z = a.x + r.x; z = (z >= 0.0f) ? z : 0.2f * z; p += z * at.x;
    z = a.y + r.y; z = (z >= 0.0f) ? z : 0.2f * z; p += z * at.y;
    z = a.z + r.z; z = (z >= 0.0f) ? z : 0.2f * z; p += z * at.z;
    z = a.w + r.w; z = (z >= 0.0f) ? z : 0.2f * z; p += z * at.w;
    // reduce within 8-lane head group
    p += __shfl_xor(p, 1, 32);
    p += __shfl_xor(p, 2, 32);
    p += __shfl_xor(p, 4, 32);
    if ((lane & 7) == 0) {
        const int h = lane >> 3;
        ebuf[(size_t)e * NHEAD + h] = p;
        atomicMax(&emax[(size_t)d * NHEAD + h], enc_f32(p));
    }
}

// E2: thread per edge: p = exp(e - emax[dst]); den[dst] += p (overwrites ebuf with p)
__global__ void __launch_bounds__(256)
edge_expsum(const long long* __restrict__ ei, const int* __restrict__ ea,
            const int* __restrict__ emax, float* __restrict__ ebuf,
            float* __restrict__ den, int rel, int nedges)
{
    const int e = blockIdx.x * 256 + threadIdx.x;
    if (e >= nedges) return;
    if (ea[e] + 1 != rel) return;
    const long long d = ei[nedges + e];
#pragma unroll
    for (int h = 0; h < NHEAD; ++h) {
        const float m = dec_f32(emax[(size_t)d * NHEAD + h]);
        const float p = expf(ebuf[(size_t)e * NHEAD + h] - m);
        ebuf[(size_t)e * NHEAD + h] = p;
        atomicAdd(&den[(size_t)d * NHEAD + h], p);
    }
}

// E3: wave per edge: acc[dst] += (p/den[dst]) * xl[src]
__global__ void __launch_bounds__(256)
edge_scatter(const long long* __restrict__ ei, const int* __restrict__ ea,
             const float* __restrict__ xl, const float* __restrict__ ebuf,
             const float* __restrict__ den, float* __restrict__ accum,
             int rel, int nedges)
{
    const int lane = threadIdx.x & 31;
    const int e    = blockIdx.x * 8 + (threadIdx.x >> 5);
    if (e >= nedges) return;
    if (ea[e] + 1 != rel) return;
    const long long s = ei[e];
    const long long d = ei[nedges + e];
    const int h  = lane >> 3;
    const float dn = den[(size_t)d * NHEAD + h];
    const float alpha = ebuf[(size_t)e * NHEAD + h] / ((dn > 0.0f) ? dn : 1.0f);
    const int c = lane << 2;
    const float4 v = *(const float4*)(xl + (size_t)s * CDIM + c);
    float* ap = accum + (size_t)d * CDIM + c;
    atomicAdd(ap + 0, alpha * v.x);
    atomicAdd(ap + 1, alpha * v.y);
    atomicAdd(ap + 2, alpha * v.z);
    atomicAdd(ap + 3, alpha * v.w);
}

// ---------------------------------------------------------------- residual + gbias
__global__ void __launch_bounds__(256)
finalize(const float* __restrict__ x, const float* __restrict__ accum,
         const float* __restrict__ gbias, float* __restrict__ out, int total)
{
    const int i = blockIdx.x * 256 + threadIdx.x;
    if (i >= total) return;
    const int c = i & (CDIM - 1);
    out[i] = x[i] + accum[i] + gbias[c] + gbias[CDIM + c] + gbias[2 * CDIM + c];
}

// ---------------------------------------------------------------- host driver
extern "C" void kernel_launch(void* const* d_in, const int* in_sizes, int n_in,
                              void* d_out, int out_size, void* d_ws, size_t ws_size,
                              hipStream_t stream)
{
    const float*      x     = (const float*)d_in[0];
    const long long*  ei    = (const long long*)d_in[1];   // int64 [2,E]
    const int*        ea    = (const int*)d_in[2];         // int32 [E]
    const float*      ln_w  = (const float*)d_in[3];
    const float*      ln_b  = (const float*)d_in[4];
    const float*      Wl    = (const float*)d_in[5];       // [3,C,C]
    const float*      bl    = (const float*)d_in[6];       // [3,C]
    const float*      Wr    = (const float*)d_in[7];
    const float*      br    = (const float*)d_in[8];
    const float*      att   = (const float*)d_in[9];       // [3,H,DH] == [3,128]
    const float*      gbias = (const float*)d_in[10];      // [3,C]
    const float*      mW1   = (const float*)d_in[11];
    const float*      mb1   = (const float*)d_in[12];
    const float*      mW2   = (const float*)d_in[13];
    const float*      mb2   = (const float*)d_in[14];
    float*            out   = (float*)d_out;

    const int N = in_sizes[0] / CDIM;   // 100000
    const int E = in_sizes[2];          // 640000
    const size_t nc = (size_t)N * CDIM;

    // workspace carve-out (256B aligned)
    char*  ws  = (char*)d_ws;
    size_t off = 0;
    auto walloc = [&](size_t bytes) -> void* {
        void* p = ws + off;
        off = (off + bytes + 255) & ~(size_t)255;
        return p;
    };
    bf16_t* x1b   = (bf16_t*)walloc(nc * sizeof(bf16_t));       // LN(x) in bf16
    bf16_t* wt    = (bf16_t*)walloc(8 * 16384 * sizeof(bf16_t)); // Wl0-2,Wr0-2,mW1,mW2 (transposed bf16)
    float*  xl    = (float*) walloc(nc * sizeof(float));
    float*  xr    = (float*) walloc(nc * sizeof(float));        // reused as bf16 h-buffer in MLP
    float*  accum = (float*) walloc(nc * sizeof(float));        // gat + mlp accumulator
    float*  ebuf  = (float*) walloc((size_t)E * NHEAD * sizeof(float));
    int*    emax  = (int*)   walloc((size_t)N * NHEAD * sizeof(int));
    float*  den   = (float*) walloc((size_t)N * NHEAD * sizeof(float));
    bf16_t* hbuf  = (bf16_t*)xr;

    // 1) weight transpose + bf16 downconvert (8 x 128x128, L2-resident afterwards)
    const float* srcW[8] = { Wl, Wl + 16384, Wl + 32768,
                             Wr, Wr + 16384, Wr + 32768, mW1, mW2 };
    for (int m = 0; m < 8; ++m)
        wt_prep<<<64, 256, 0, stream>>>(srcW[m], wt + (size_t)m * 16384);

    // 2) LayerNorm -> bf16
    ln_to_bf16<<<(N + 7) / 8, 256, 0, stream>>>(x, ln_w, ln_b, x1b, N);

    // 3) zero the GAT/MLP accumulator
    fill_f32<<<(int)((nc + 255) / 256), 256, 0, stream>>>(accum, 0.0f, (int)nc);

    const int rowTiles = (N + 15) / 16;
    const int gemmGrid = (rowTiles + 7) / 8;
    const int edgeWGrid = (E + 7) / 8;      // wave-per-edge kernels
    const int edgeTGrid = (E + 255) / 256;  // thread-per-edge kernel
    const int nhTotal = N * NHEAD;

    // 4) per-relation GATv2
    for (int r = 0; r < 3; ++r) {
        gemm_bf16_wmma<<<gemmGrid, 256, 0, stream>>>(
            x1b, wt + (size_t)r * 16384, bl + r * CDIM, xl, nullptr, N, 0, 0);
        gemm_bf16_wmma<<<gemmGrid, 256, 0, stream>>>(
            x1b, wt + (size_t)(3 + r) * 16384, br + r * CDIM, xr, nullptr, N, 0, 0);

        fill_i32<<<(nhTotal + 255) / 256, 256, 0, stream>>>(emax, (int)0x80000000, nhTotal);
        fill_f32<<<(nhTotal + 255) / 256, 256, 0, stream>>>(den, 0.0f, nhTotal);

        edge_logits<<<edgeWGrid, 256, 0, stream>>>(ei, ea, xl, xr, att + r * CDIM,
                                                   ebuf, emax, r, E);
        edge_expsum<<<edgeTGrid, 256, 0, stream>>>(ei, ea, emax, ebuf, den, r, E);
        edge_scatter<<<edgeWGrid, 256, 0, stream>>>(ei, ea, xl, ebuf, den, accum, r, E);
    }

    // 5) MLP: h = gelu(x1 @ mW1 + mb1) -> bf16 ; accum += gelu(h @ mW2 + mb2)
    gemm_bf16_wmma<<<gemmGrid, 256, 0, stream>>>(
        x1b, wt + 6 * 16384, mb1, nullptr, hbuf, N, 1, 1);
    gemm_bf16_wmma<<<gemmGrid, 256, 0, stream>>>(
        hbuf, wt + 7 * 16384, mb2, accum, nullptr, N, 1, 2);

    // 6) out = x + gat + mlp + sum_r gbias[r]
    finalize<<<(int)((nc + 255) / 256), 256, 0, stream>>>(x, accum, gbias, out, (int)nc);
}
